// SMRNSSMOnly_4011499454833
// MI455X (gfx1250) — matrix operand order repaired and checked by reading, not compiled, not run on hardware
//
#include <hip/hip_runtime.h>
#include <hip/hip_bf16.h>

typedef _Float16 half_t;
typedef __attribute__((ext_vector_type(16))) _Float16 v16h;
typedef __attribute__((ext_vector_type(8)))  _Float16 v8h;
typedef __attribute__((ext_vector_type(8)))  float    v8f;

#define kVOCAB 16000
#define kD     512
#define kFD    2048
#define kL     4
#define kN     16
#define kT     1024
#define kB     4
#define kR     32
#define kM     (kB*kT)   /* 4096 token rows */

union ABFrag { v16h v; v8h h[2]; };

// ---------------------------------------------------------------------------
// fp32 -> f16 conversion kernels (weights done once per launch; cheap)
// ---------------------------------------------------------------------------
__global__ void k_cvt(const float* __restrict__ s, half_t* __restrict__ d, int n) {
  for (int i = blockIdx.x * blockDim.x + threadIdx.x; i < n; i += gridDim.x * blockDim.x)
    d[i] = (half_t)s[i];
}

// w1T[l][j][d] = ffn_w1[l][d][j]   (so B-operand rows are K(=d)-contiguous)
__global__ void k_cvt_w1T(const float* __restrict__ w1, half_t* __restrict__ w1T, int total) {
  for (int i = blockIdx.x * blockDim.x + threadIdx.x; i < total; i += gridDim.x * blockDim.x) {
    int l = i / (kFD * kD); int r = i % (kFD * kD); int j = r / kD; int dd = r % kD;
    w1T[i] = (half_t)w1[((size_t)l * kD + dd) * kFD + j];
  }
}

// w2T[l][d][j] = ffn_w2[l][j][d]   (B-operand rows are K(=j)-contiguous)
__global__ void k_cvt_w2T(const float* __restrict__ w2, half_t* __restrict__ w2T, int total) {
  for (int i = blockIdx.x * blockDim.x + threadIdx.x; i < total; i += gridDim.x * blockDim.x) {
    int l = i / (kD * kFD); int r = i % (kD * kFD); int dd = r / kFD; int j = r % kFD;
    w2T[i] = (half_t)w2[((size_t)l * kFD + j) * kD + dd];
  }
}

// ---------------------------------------------------------------------------
// x = embed[tokens] + pos
// ---------------------------------------------------------------------------
__global__ void k_embed(const int* __restrict__ tok, const float* __restrict__ emb,
                        const float* __restrict__ pos, float* __restrict__ xF) {
  for (int i = blockIdx.x * blockDim.x + threadIdx.x; i < kM * kD; i += gridDim.x * blockDim.x) {
    int bt = i / kD, dd = i % kD;
    int t = bt % kT;
    xF[i] = emb[(size_t)tok[bt] * kD + dd] + pos[(size_t)t * kD + dd];
  }
}

// ---------------------------------------------------------------------------
// xp = x @ xproj_w (48 cols)  and  C = x @ C_w (16 cols); fp32, LDS row reuse
// ---------------------------------------------------------------------------
__global__ __launch_bounds__(64)
void k_xproj(const float* __restrict__ xF, const float* __restrict__ xw,
             const float* __restrict__ cw, float* __restrict__ xpF, float* __restrict__ cF) {
  int m = blockIdx.x;
  int c = threadIdx.x;                  // 64 threads: 48 xproj cols + 16 C cols
  __shared__ float xs[kD];
  for (int i = c; i < kD; i += 64) xs[i] = xF[(size_t)m * kD + i];
  __syncthreads();
  float acc = 0.f;
  if (c < kR + kN) {                    // 48 xproj columns
    for (int dd = 0; dd < kD; ++dd) acc += xs[dd] * xw[dd * (kR + kN) + c];
    xpF[(size_t)m * (kR + kN) + c] = acc;
  } else {                              // 16 C columns
    int n = c - (kR + kN);
    for (int dd = 0; dd < kD; ++dd) acc += xs[dd] * cw[dd * kN + n];
    cF[(size_t)m * kN + n] = acc;
  }
}

// ---------------------------------------------------------------------------
// delta = softplus(dlow @ dt_w + dt_b); fp32
// ---------------------------------------------------------------------------
__global__ __launch_bounds__(256)
void k_delta(const float* __restrict__ xpF, const float* __restrict__ dtw,
             const float* __restrict__ dtb, float* __restrict__ dF) {
  int m = blockIdx.x;
  int tid = threadIdx.x;
  __shared__ float dl[kR];
  if (tid < kR) dl[tid] = xpF[(size_t)m * (kR + kN) + tid];
  __syncthreads();
  for (int c = tid; c < kD; c += 256) {
    float acc = dtb[c];
    for (int r = 0; r < kR; ++r) acc += dl[r] * dtw[r * kD + c];
    dF[(size_t)m * kD + c] = fmaxf(acc, 0.f) + log1pf(expf(-fabsf(acc)));  // stable softplus
  }
}

// ---------------------------------------------------------------------------
// Selective scan, lane-parallel over the state dimension:
// one lane per (d, n) pair; n = lane&15, so y = sum_n C[n]*h[n] is a 4-step
// __shfl_xor reduction inside a half-wave. No LDS, no barriers, 1024 waves.
// ---------------------------------------------------------------------------
__global__ __launch_bounds__(256)
void k_scan(const float* __restrict__ xF, const float* __restrict__ xpF,
            const float* __restrict__ cF, const float* __restrict__ dF,
            const float* __restrict__ A_log, float* __restrict__ yF) {
  const int tid  = threadIdx.x;
  const int n    = tid & 15;            // state index within half-wave
  const int dsub = tid >> 4;            // 0..15: d-channel group within block
  const int b    = blockIdx.x / (kD / 16);
  const int d    = (blockIdx.x % (kD / 16)) * 16 + dsub;

  const float a = -expf(A_log[d * kN + n]);
  float h = 0.f;
  for (int t = 0; t < kT; ++t) {
    const size_t row = (size_t)b * kT + t;
    const float dlt = dF[row * kD + d];                    // broadcast across 16 lanes
    const float xt  = xF[row * kD + d];                    // broadcast across 16 lanes
    const float Bn  = xpF[row * (kR + kN) + kR + n];
    const float Cn  = cF[row * kN + n];
    h = expf(dlt * a) * h + (dlt * xt) * Bn;
    float p = Cn * h;
    p += __shfl_xor(p, 1, 32);
    p += __shfl_xor(p, 2, 32);
    p += __shfl_xor(p, 4, 32);
    p += __shfl_xor(p, 8, 32);
    if (n == 0) yF[row * kD + d] = p;
  }
}

// ---------------------------------------------------------------------------
// LayerNorm (two-pass), optional residual add; writes fp32 and/or f16
// ---------------------------------------------------------------------------
__global__ __launch_bounds__(256)
void k_ln(const float* __restrict__ xin, const float* __restrict__ yin,
          const float* __restrict__ g, const float* __restrict__ bb,
          float* __restrict__ outF, half_t* __restrict__ outH) {
  int m = blockIdx.x;
  int tid = threadIdx.x;
  __shared__ float red[8];
  size_t base = (size_t)m * kD;
  float v0 = xin[base + tid]       + (yin ? yin[base + tid]       : 0.f);
  float v1 = xin[base + tid + 256] + (yin ? yin[base + tid + 256] : 0.f);
  float s = v0 + v1;
  for (int o = 16; o > 0; o >>= 1) s += __shfl_down(s, o, 32);
  if ((tid & 31) == 0) red[tid >> 5] = s;
  __syncthreads();
  float tot = 0.f;
  for (int i = 0; i < 8; ++i) tot += red[i];
  float mu = tot * (1.f / kD);
  __syncthreads();
  float d0 = v0 - mu, d1 = v1 - mu;
  float q = d0 * d0 + d1 * d1;
  for (int o = 16; o > 0; o >>= 1) q += __shfl_down(q, o, 32);
  if ((tid & 31) == 0) red[tid >> 5] = q;
  __syncthreads();
  float qt = 0.f;
  for (int i = 0; i < 8; ++i) qt += red[i];
  float inv = rsqrtf(qt * (1.f / kD) + 1e-5f);
  float o0 = d0 * inv * g[tid]       + bb[tid];
  float o1 = d1 * inv * g[tid + 256] + bb[tid + 256];
  if (outF) { outF[base + tid] = o0; outF[base + tid + 256] = o1; }
  if (outH) { outH[base + tid] = (half_t)o0; outH[base + tid + 256] = (half_t)o1; }
}

// ---------------------------------------------------------------------------
// WMMA f16 GEMM: C(MxN) = A(MxK, row-major f16) * Bt(NxK, row-major f16)^T
// Block = 256 threads = 8 waves; 64x64 tile; wave = 16x32 (two 16x16 frags).
// MODE 0: outF = acc            (logits)
// MODE 1: outH = f16(gelu(acc + bias))   (FFN1)
// MODE 2: outF += acc + bias    (FFN2 with residual already in outF)
// ---------------------------------------------------------------------------
template <int MODE>
__global__ __launch_bounds__(256)
void k_gemm_wmma(const half_t* __restrict__ A, const half_t* __restrict__ Bt,
                 const float* __restrict__ bias, float* __restrict__ outF,
                 half_t* __restrict__ outH, int M, int N, int K) {
  const int lane = threadIdx.x & 31;
  const int wave = threadIdx.x >> 5;
  const int l15  = lane & 15;
  const int hf   = lane >> 4;          // 0 or 1 (lane half)
  const int wm   = wave & 3;           // 4 waves along M
  const int wn   = wave >> 2;          // 2 waves along N
  const int m0   = blockIdx.y * 64 + wm * 16;
  const int n0   = blockIdx.x * 64 + wn * 32;

  v8f acc[2] = {v8f{}, v8f{}};
  const half_t* arow  = A  + (size_t)(m0 + l15) * K;
  const half_t* brow0 = Bt + (size_t)(n0 + l15) * K;
  const half_t* brow1 = Bt + (size_t)(n0 + 16 + l15) * K;

  for (int k0 = 0; k0 < K; k0 += 32) {
    ABFrag a, b0, b1;
    // A 16x32 f16, contiguous k order used here (same as B): lanes 0-15 hold K 0..15, lanes 16-31 hold K 16..31
    a.h[0]  = *(const v8h*)(arow + k0 + hf * 16);
    a.h[1]  = *(const v8h*)(arow + k0 + hf * 16 + 8);
    // B 32x16 f16, contiguous k order used here (same as A): lanes 0-15 hold K 0..15, lanes 16-31 hold K 16..31
    b0.h[0] = *(const v8h*)(brow0 + k0 + hf * 16);
    b0.h[1] = *(const v8h*)(brow0 + k0 + hf * 16 + 8);
    b1.h[0] = *(const v8h*)(brow1 + k0 + hf * 16);
    b1.h[1] = *(const v8h*)(brow1 + k0 + hf * 16 + 8);
    __builtin_prefetch(arow + k0 + 128, 0, 1);
    __builtin_prefetch(brow0 + k0 + 128, 0, 1);
    acc[0] = __builtin_amdgcn_wmma_f32_16x16x32_f16(false, a.v, false, b0.v,
                                                    (short)0, acc[0], false, false);
    acc[1] = __builtin_amdgcn_wmma_f32_16x16x32_f16(false, a.v, false, b1.v,
                                                    (short)0, acc[1], false, false);
  }

#pragma unroll
  for (int r = 0; r < 8; ++r) {
    int m = m0 + r + 8 * hf;           // C/D layout: vgpr r, lane half hf
#pragma unroll
    for (int s = 0; s < 2; ++s) {
      int n = n0 + s * 16 + l15;
      float v = acc[s][r];
      size_t o = (size_t)m * N + n;
      if (MODE == 0) {
        outF[o] = v;
      } else if (MODE == 1) {
        v += bias[n];
        v = 0.5f * v * (1.0f + erff(v * 0.7071067811865475f));  // exact gelu
        outH[o] = (half_t)v;
      } else {
        outF[o] += v + bias[n];        // residual already resident in outF
      }
    }
  }
}

// ---------------------------------------------------------------------------
extern "C" void kernel_launch(void* const* d_in, const int* in_sizes, int n_in,
                              void* d_out, int out_size, void* d_ws, size_t ws_size,
                              hipStream_t stream) {
  const int*   tokens  = (const int*)d_in[0];
  const float* embed   = (const float*)d_in[1];
  const float* pos     = (const float*)d_in[2];
  const float* A_log   = (const float*)d_in[3];
  const float* xproj_w = (const float*)d_in[4];
  const float* dt_w    = (const float*)d_in[5];
  const float* dt_b    = (const float*)d_in[6];
  const float* C_w     = (const float*)d_in[7];
  const float* ln_g    = (const float*)d_in[8];
  const float* ln_b    = (const float*)d_in[9];
  const float* ffn_w1  = (const float*)d_in[10];
  const float* ffn_b1  = (const float*)d_in[11];
  const float* ffn_w2  = (const float*)d_in[12];
  const float* ffn_b2  = (const float*)d_in[13];
  const float* lnout_g = (const float*)d_in[14];
  const float* lnout_b = (const float*)d_in[15];
  float* out = (float*)d_out;
  (void)in_sizes; (void)n_in; (void)out_size; (void)ws_size;

  char* base = (char*)d_ws;
  size_t off = 0;
  auto carve = [&](size_t bytes) -> char* {
    char* p = base + off;
    off += (bytes + 255) & ~(size_t)255;
    return p;
  };
  float*  xF   = (float*)carve(sizeof(float) * kM * kD);
  float*  yF   = (float*)carve(sizeof(float) * kM * kD);
  float*  dF   = (float*)carve(sizeof(float) * kM * kD);
  float*  xpF  = (float*)carve(sizeof(float) * kM * (kR + kN));
  float*  cF   = (float*)carve(sizeof(float) * kM * kN);
  half_t* xH   = (half_t*)carve(sizeof(half_t) * kM * kD);
  half_t* hH   = (half_t*)carve(sizeof(half_t) * kM * kFD);
  half_t* w1T  = (half_t*)carve(sizeof(half_t) * kL * kFD * kD);
  half_t* w2T  = (half_t*)carve(sizeof(half_t) * kL * kD * kFD);
  half_t* embH = (half_t*)carve(sizeof(half_t) * kVOCAB * kD);

  // one-time weight conversion / transposition to f16
  {
    int n = kVOCAB * kD;
    k_cvt<<<(n + 255) / 256, 256, 0, stream>>>(embed, embH, n);
    int t1 = kL * kFD * kD;
    k_cvt_w1T<<<(t1 + 255) / 256, 256, 0, stream>>>(ffn_w1, w1T, t1);
    k_cvt_w2T<<<(t1 + 255) / 256, 256, 0, stream>>>(ffn_w2, w2T, t1);
  }

  // x = embed[tokens] + pos
  k_embed<<<(kM * kD + 255) / 256, 256, 0, stream>>>(tokens, embed, pos, xF);

  for (int l = 0; l < kL; ++l) {
    k_xproj<<<kM, 64, 0, stream>>>(xF, xproj_w + (size_t)l * kD * (kR + kN),
                                   C_w + (size_t)l * kD * kN, xpF, cF);
    k_delta<<<kM, 256, 0, stream>>>(xpF, dt_w + (size_t)l * kR * kD,
                                    dt_b + (size_t)l * kD, dF);
    // lane-parallel scan: kB * kD/16 blocks x 256 threads (one lane per (d,n))
    k_scan<<<kB * (kD / 16), 256, 0, stream>>>(xF, xpF, cF, dF,
                                               A_log + (size_t)l * kD * kN, yF);
    // x = LN(x + y) in place; also emit f16 copy for WMMA
    k_ln<<<kM, 256, 0, stream>>>(xF, yF, ln_g + (size_t)l * kD, ln_b + (size_t)l * kD, xF, xH);
    // h = gelu(x @ W1 + b1) -> f16
    k_gemm_wmma<1><<<dim3(kFD / 64, kM / 64), 256, 0, stream>>>(
        xH, w1T + (size_t)l * kFD * kD, ffn_b1 + (size_t)l * kFD, nullptr, hH, kM, kFD, kD);
    // x += h @ W2 + b2 (fp32 residual accumulation)
    k_gemm_wmma<2><<<dim3(kD / 64, kM / 64), 256, 0, stream>>>(
        hH, w2T + (size_t)l * kD * kFD, ffn_b2 + (size_t)l * kD, xF, nullptr, kM, kD, kFD);
  }

  // final layernorm -> f16
  k_ln<<<kM, 256, 0, stream>>>(xF, nullptr, lnout_g, lnout_b, nullptr, xH);
  // logits = x @ embed^T  (embed rows are already K-contiguous)
  k_gemm_wmma<0><<<dim3(kVOCAB / 64, kM / 64), 256, 0, stream>>>(
      xH, embH, nullptr, out, nullptr, kM, kVOCAB, kD);
}
